// DeepGainModel_28707561407391
// MI455X (gfx1250) — matrix operand order repaired and checked by reading, not compiled, not run on hardware
//
#include <hip/hip_runtime.h>
#include <math.h>

// ---------------------------------------------------------------------------
// DeepGainModel scan for MI455X (gfx1250), wave32, WMMA f32_16x16x32_f16.
//
// Pipeline (all on `stream`):
//   P1  precompute_tables : fold e_emb/mu_emb/bias through layer-1 weights
//   P2  swizzle_bfrag x4  : fp32 weights -> fp16 WMMA B-fragment order
//   K1  scan_kernel       : sequential T-loop, FatigueNet via LDS-resident
//                           WMMA weights, writes mpc trajectory + final mpc
//   K2  rir_kernel        : fully parallel RIRNet over all B*T rows (WMMA)
// ---------------------------------------------------------------------------

typedef _Float16 v16h __attribute__((ext_vector_type(16)));
typedef _Float16 v8h  __attribute__((ext_vector_type(8)));
typedef float    v8f  __attribute__((ext_vector_type(8)));

#define BB   256
#define TT   128
#define MM   16
#define NEX  27
#define HH   256
#define DT_SCALE 5.1298987149230735f   // log1p(168)

// ---- LDS layout (bytes). H/W-frag offsets shared by both WMMA kernels. ----
#define LD1   264      // halves per row of h1/h2 (256 + 8 pad -> conflict-free)
#define LD3   136      // halves per row of h3   (128 + 8 pad)
#define OFF_W2F  0          // 131072 B : layer2 weights, fragment order
#define OFF_W3F  131072     // 65536  B : layer3 weights, fragment order
#define OFF_H1   196608     // 64*264*2 = 33792 B
#define OFF_H2   230400     // 33792 B
#define OFF_H3   264192     // 64*136*2 = 17408 B
// scan_kernel extras
#define OFF_FR   281600     // 4*256 f32: fW1 rows {w,rp,rt,mpc}
#define OFF_FB2  285696     // 256 f32
#define OFF_FB3  286720     // 128 f32
#define OFF_FW4  287232     // 128 f32
#define OFF_TAU  287744     // 16 f32
#define OFF_INV  287808     // 27*16 f32
#define OFF_MPC  289536     // 64 f32 (state)
#define OFF_SCL  289792     // 12 f32 (w,rp,rt per local batch)
#define OFF_EIX  289856     // 4 int
#define OFF_FMU  289920     // 16*256 f32: mu_emb@fW1 + fb1
#define OFF_EROW 306304     // 4*256 f32: fE rows for this step's exercises
#define SMEM_B   310400
// rir_kernel extras
#define OFF_G1M  281600     // 16*256 f32: gW1 mpc block
#define OFF_GRW  297984     // 2*256 f32: gW1 rows {w,rp}
#define OFF_GB2  300032     // 256 f32
#define OFF_GB3  301056     // 128 f32
#define OFF_GW4  301568     // 128 f32
#define OFF_MPCC 302080     // 64*16 f32
#define OFF_SW   306176     // 64 f32
#define OFF_SRP  306432     // 64 f32
#define OFF_EIXR 306688     // 64 int
#define SMEM_C   306944

// ---- workspace layout (bytes) ----
#define WS_MPC   0                         // B*T*M f32 = 2097152
#define WS_FE    2097152                   // 27*256 f32
#define WS_FMU   2124800                   // 16*256 f32
#define WS_GE    2141184                   // 27*256 f32
#define WS_W2F   2168832                   // 131072 (f16)
#define WS_W3F   2299904                   // 65536
#define WS_GW2F  2365440                   // 131072
#define WS_GW3F  2496512                   // 65536

__device__ __forceinline__ void cp_lds(void* dst, const void* src, int bytes,
                                       int tid, int nthr) {
  uint4* d = (uint4*)dst;
  const uint4* s = (const uint4*)src;
  int n = bytes >> 4;
  for (int i = tid; i < n; i += nthr) d[i] = s[i];
}

// ---------------------------------------------------------------------------
// P1: fE[e][h] = ex_emb[e]@fW1_e ; fMuB[m][h] = mu_emb[m]@fW1_mu + fb1 ;
//     gE[e][h] = ex_emb[e]@gW1_e + gb1
// fx layout rows: 0=w 1=rp 2=rt 3=mpc 4..67=e_emb 68..131=mu_emb
// gx layout rows: 0=w 1=rp 2..65=e_emb 66..81=mpc
// ---------------------------------------------------------------------------
__global__ __launch_bounds__(256) void precompute_tables(
    const float* __restrict__ exe, const float* __restrict__ mue,
    const float* __restrict__ fW1, const float* __restrict__ fb1,
    const float* __restrict__ gW1, const float* __restrict__ gb1,
    float* __restrict__ fE, float* __restrict__ fMuB, float* __restrict__ gE) {
  int h = threadIdx.x;
  int b = blockIdx.x;
  if (b < NEX) {
    float acc = 0.f;
    for (int j = 0; j < 64; ++j) acc += exe[b * 64 + j] * fW1[(4 + j) * HH + h];
    fE[b * HH + h] = acc;
  } else if (b < NEX + MM) {
    int m = b - NEX;
    float acc = fb1[h];
    for (int j = 0; j < 64; ++j) acc += mue[m * 64 + j] * fW1[(68 + j) * HH + h];
    fMuB[m * HH + h] = acc;
  } else {
    int e = b - NEX - MM;
    float acc = gb1[h];
    for (int j = 0; j < 64; ++j) acc += exe[e * 64 + j] * gW1[(2 + j) * HH + h];
    gE[e * HH + h] = acc;
  }
}

// ---------------------------------------------------------------------------
// P2: swizzle fp32 row-major W[K][N] into fp16 WMMA B-fragment order:
// frag[((kk*NT+nt)*32+lane)*16 + j] = W[kk*32 + 8*(lane>>4) + (j&7) + 16*(j>>3)][nt*16 + lane&15]
// (B 32x16: lanes 0-15 K=0..7,16..23, lanes 16-31 K=8..15,24..31, matching A.)
// ---------------------------------------------------------------------------
__global__ __launch_bounds__(256) void swizzle_bfrag(
    const float* __restrict__ W, _Float16* __restrict__ frag, int NT, int ldW) {
  int kk = blockIdx.x / NT, nt = blockIdx.x % NT;
  for (int idx = threadIdx.x; idx < 512; idx += 256) {
    int lane = idx >> 4, j = idx & 15;
    int g = lane >> 4, nl = lane & 15;
    frag[(((kk * NT + nt) * 32) + lane) * 16 + j] =
        (_Float16)W[(kk * 32 + g * 8 + (j & 7) + (j >> 3) * 16) * ldW + nt * 16 + nl];
  }
}

// ---------------------------------------------------------------------------
// Shared WMMA middle: h2 = relu(h1@W2+b2)  (64x256 @ 256x256),
//                     h3 = relu(h2@W3+b3)  (64x256 @ 256x128)
// Weights are LDS-resident fragments at OFF_W2F/OFF_W3F.
// ---------------------------------------------------------------------------
__device__ __forceinline__ void mlp_mid(char* smem, const float* __restrict__ b2,
                                        const float* __restrict__ b3, int tid) {
  const _Float16* sH1 = (const _Float16*)(smem + OFF_H1);
  _Float16* sH2 = (_Float16*)(smem + OFF_H2);
  _Float16* sH3 = (_Float16*)(smem + OFF_H3);
  const _Float16* w2f = (const _Float16*)(smem + OFF_W2F);
  const _Float16* w3f = (const _Float16*)(smem + OFF_W3F);
  const int lane = tid & 31, wv = tid >> 5;
  const int g = lane >> 4, nl = lane & 15;
  __syncthreads();  // h1 ready
  // ---- layer 2: 4 row-tiles x 16 col-tiles, 8 waves -> 8 tiles each ----
  for (int i = 0; i < 8; ++i) {
    int tIdx = wv * 8 + i;
    int rt = tIdx >> 4, nt = tIdx & 15;
    v8f acc = {};
    const _Float16* rowp = sH1 + (rt * 16 + nl) * LD1 + g * 8;
#pragma unroll
    for (int kk = 0; kk < 8; ++kk) {
      // A 16x32 f16: lane nl = row, K = kbase + j + 8g (j<8) / +16 (j>=8)
      v8h a0 = *(const v8h*)(rowp + kk * 32);
      v8h a1 = *(const v8h*)(rowp + kk * 32 + 16);
      v16h av = __builtin_shufflevector(a0, a1, 0, 1, 2, 3, 4, 5, 6, 7, 8, 9,
                                        10, 11, 12, 13, 14, 15);
      const _Float16* pb = w2f + (((kk * 16 + nt) * 32 + lane) << 4);
      v8h b0 = *(const v8h*)pb;
      v8h b1 = *(const v8h*)(pb + 8);
      v16h bv = __builtin_shufflevector(b0, b1, 0, 1, 2, 3, 4, 5, 6, 7, 8, 9,
                                        10, 11, 12, 13, 14, 15);
      acc = __builtin_amdgcn_wmma_f32_16x16x32_f16(false, av, false, bv,
                                                   (short)0, acc, false, false);
    }
#pragma unroll
    for (int vr = 0; vr < 8; ++vr) {  // D: M = vr + 8g, N = nl
      int row = rt * 16 + g * 8 + vr;
      int col = nt * 16 + nl;
      sH2[row * LD1 + col] = (_Float16)fmaxf(acc[vr] + b2[col], 0.f);
    }
  }
  __syncthreads();
  // ---- layer 3: 4 x 8 tiles, 8 waves -> 4 tiles each ----
  for (int i = 0; i < 4; ++i) {
    int tIdx = wv * 4 + i;
    int rt = tIdx >> 3, nt = tIdx & 7;
    v8f acc = {};
    const _Float16* rowp = sH2 + (rt * 16 + nl) * LD1 + g * 8;
#pragma unroll
    for (int kk = 0; kk < 8; ++kk) {
      v8h a0 = *(const v8h*)(rowp + kk * 32);
      v8h a1 = *(const v8h*)(rowp + kk * 32 + 16);
      v16h av = __builtin_shufflevector(a0, a1, 0, 1, 2, 3, 4, 5, 6, 7, 8, 9,
                                        10, 11, 12, 13, 14, 15);
      const _Float16* pb = w3f + (((kk * 8 + nt) * 32 + lane) << 4);
      v8h b0 = *(const v8h*)pb;
      v8h b1 = *(const v8h*)(pb + 8);
      v16h bv = __builtin_shufflevector(b0, b1, 0, 1, 2, 3, 4, 5, 6, 7, 8, 9,
                                        10, 11, 12, 13, 14, 15);
      acc = __builtin_amdgcn_wmma_f32_16x16x32_f16(false, av, false, bv,
                                                   (short)0, acc, false, false);
    }
#pragma unroll
    for (int vr = 0; vr < 8; ++vr) {
      int row = rt * 16 + g * 8 + vr;
      int col = nt * 16 + nl;
      sH3[row * LD3 + col] = (_Float16)fmaxf(acc[vr] + b3[col], 0.f);
    }
  }
  __syncthreads();  // h3 ready for layer-4 dot
}

// ---------------------------------------------------------------------------
// K1: sequential scan. Block owns 4 batch rows (R = 64 FatigueNet rows).
// Layer-2/3 weights live in LDS for the whole T loop.
// ---------------------------------------------------------------------------
__global__ __launch_bounds__(256) void scan_kernel(
    const int* __restrict__ eidx, const float* __restrict__ wgt,
    const float* __restrict__ rps, const float* __restrict__ rrt,
    const float* __restrict__ dlt, const float* __restrict__ log_tau,
    const float* __restrict__ involvement, const float* __restrict__ fb2,
    const float* __restrict__ fb3, const float* __restrict__ fW4,
    const float* __restrict__ fb4, const float* __restrict__ fW1,
    const float* __restrict__ fE, const float* __restrict__ fMuB,
    const _Float16* __restrict__ w2frag, const _Float16* __restrict__ w3frag,
    float* __restrict__ mpcPost, float* __restrict__ outMpc) {
  extern __shared__ char smem[];
  const int tid = threadIdx.x;
  cp_lds(smem + OFF_W2F, w2frag, 131072, tid, 256);
  cp_lds(smem + OFF_W3F, w3frag, 65536, tid, 256);
  cp_lds(smem + OFF_FR, fW1, 4096, tid, 256);   // rows 0..3 of fW1
  cp_lds(smem + OFF_FB2, fb2, 1024, tid, 256);
  cp_lds(smem + OFF_FB3, fb3, 512, tid, 256);
  cp_lds(smem + OFF_FW4, fW4, 512, tid, 256);
  cp_lds(smem + OFF_INV, involvement, 1728, tid, 256);
  cp_lds(smem + OFF_FMU, fMuB, 16384, tid, 256);
  float* sTau = (float*)(smem + OFF_TAU);
  float* sMpc = (float*)(smem + OFF_MPC);
  float* sScal = (float*)(smem + OFF_SCL);
  int* sEix = (int*)(smem + OFF_EIX);
  float* sERow = (float*)(smem + OFF_EROW);
  const float* sFr = (const float*)(smem + OFF_FR);
  const float* sFMu = (const float*)(smem + OFF_FMU);
  _Float16* sH1 = (_Float16*)(smem + OFF_H1);
  if (tid < MM) sTau[tid] = expf(log_tau[tid]);
  if (tid < 64) sMpc[tid] = 1.0f;
  __syncthreads();

  for (int t = 0; t < TT; ++t) {
    if (tid < 4) {  // per-batch scalars for this step
      int bg = blockIdx.x * 4 + tid;
      int off = bg * TT + t;
      sScal[tid] = wgt[off];
      sScal[4 + tid] = rps[off];
      sScal[8 + tid] = rrt[off];
      sEix[tid] = eidx[off];
    }
    if (tid < 64) {  // ExponentialRecovery (skipped at t=0)
      int b = tid >> 4, m = tid & 15;
      int bg = blockIdx.x * 4 + b;
      float mpc = sMpc[tid];
      float dth = expm1f(dlt[bg * TT + t] * DT_SCALE);
      float rec = 1.f - (1.f - mpc) * expf(-dth / sTau[m]);
      if (t > 0) mpc = rec;
      sMpc[tid] = mpc;
      mpcPost[(bg * TT + t) * MM + m] = mpc;  // consumed by rir_kernel
    }
    // stage this step's 4 exercise projection rows (uniform e per batch row)
    for (int bi = 0; bi < 4; ++bi) {
      int e = eidx[(blockIdx.x * 4 + bi) * TT + t];
      sERow[bi * HH + tid] = fE[e * HH + tid];
    }
    __syncthreads();
    // h1[r][h] = relu(table sums + 4 scalar FMAs), r = 16*b + m
    for (int r = 0; r < 64; ++r) {
      int h = tid, b = r >> 4, m = r & 15;
      float v = sFMu[m * HH + h] + sERow[b * HH + h] + sScal[b] * sFr[h] +
                sScal[4 + b] * sFr[256 + h] + sScal[8 + b] * sFr[512 + h] +
                sMpc[r] * sFr[768 + h];
      sH1[r * LD1 + h] = (_Float16)fmaxf(v, 0.f);
    }
    mlp_mid(smem, (const float*)(smem + OFF_FB2), (const float*)(smem + OFF_FB3),
            tid);
    if (tid < 64) {  // layer 4 + sigmoid + state update
      float acc = fb4[0];
      const _Float16* hr = (const _Float16*)(smem + OFF_H3) + tid * LD3;
      const float* w4 = (const float*)(smem + OFF_FW4);
      for (int k = 0; k < 128; ++k) acc += (float)hr[k] * w4[k];
      float dropv = 1.f / (1.f + expf(-acc));
      int b = tid >> 4, m = tid & 15;
      const float* inv = (const float*)(smem + OFF_INV);
      float iv = inv[sEix[b] * MM + m];
      sMpc[tid] = fmaxf(sMpc[tid] * (1.f - iv * dropv), 0.1f);
    }
    __syncthreads();
  }
  if (tid < 64) {
    int b = tid >> 4, m = tid & 15;
    outMpc[(blockIdx.x * 4 + b) * MM + m] = sMpc[tid];
  }
}

// ---------------------------------------------------------------------------
// K2: RIRNet over all B*T rows (rir never feeds back into the scan).
// 64 rows per block, 512 blocks.
// ---------------------------------------------------------------------------
__global__ __launch_bounds__(256) void rir_kernel(
    const int* __restrict__ eidx, const float* __restrict__ wgt,
    const float* __restrict__ rps, const float* __restrict__ gW1,
    const float* __restrict__ gb2, const float* __restrict__ gb3,
    const float* __restrict__ gW4, const float* __restrict__ gb4,
    const float* __restrict__ gE, const _Float16* __restrict__ gw2frag,
    const _Float16* __restrict__ gw3frag, const float* __restrict__ mpcPost,
    float* __restrict__ outRir) {
  extern __shared__ char smem[];
  const int tid = threadIdx.x;
  const int base = blockIdx.x * 64;  // first (b,t) flat row
  cp_lds(smem + OFF_W2F, gw2frag, 131072, tid, 256);
  cp_lds(smem + OFF_W3F, gw3frag, 65536, tid, 256);
  cp_lds(smem + OFF_G1M, gW1 + 66 * HH, 16384, tid, 256);  // mpc block
  cp_lds(smem + OFF_GRW, gW1, 2048, tid, 256);             // rows w, rp
  cp_lds(smem + OFF_GB2, gb2, 1024, tid, 256);
  cp_lds(smem + OFF_GB3, gb3, 512, tid, 256);
  cp_lds(smem + OFF_GW4, gW4, 512, tid, 256);
  cp_lds(smem + OFF_MPCC, mpcPost + base * MM, 4096, tid, 256);
  float* sSw = (float*)(smem + OFF_SW);
  float* sRp = (float*)(smem + OFF_SRP);
  int* sEix = (int*)(smem + OFF_EIXR);
  const float* sMpc = (const float*)(smem + OFF_MPCC);
  const float* sG1m = (const float*)(smem + OFF_G1M);
  const float* sGr = (const float*)(smem + OFF_GRW);
  _Float16* sH1 = (_Float16*)(smem + OFF_H1);
  if (tid < 64) {
    int rg = base + tid;
    int off = (rg >> 7) * TT + (rg & 127);  // == rg (T=128), kept explicit
    sSw[tid] = wgt[off];
    sRp[tid] = rps[off];
    sEix[tid] = eidx[off];
  }
  __syncthreads();
  // h1[r][h] = relu(gE[e] + w*row0 + rp*row1 + mpc(16) . gW1_mpc)
  for (int r = 0; r < 64; ++r) {
    int h = tid;
    float v = gE[sEix[r] * HH + h] + sSw[r] * sGr[h] + sRp[r] * sGr[256 + h];
#pragma unroll
    for (int m = 0; m < 16; ++m) v += sMpc[r * 16 + m] * sG1m[m * HH + h];
    sH1[r * LD1 + h] = (_Float16)fmaxf(v, 0.f);
  }
  mlp_mid(smem, (const float*)(smem + OFF_GB2), (const float*)(smem + OFF_GB3),
          tid);
  if (tid < 64) {
    float acc = gb4[0];
    const _Float16* hr = (const _Float16*)(smem + OFF_H3) + tid * LD3;
    const float* w4 = (const float*)(smem + OFF_GW4);
    for (int k = 0; k < 128; ++k) acc += (float)hr[k] * w4[k];
    outRir[base + tid] = 1.f / (1.f + expf(-acc));  // rirs.T flat (B,T)
  }
}

// ---------------------------------------------------------------------------
extern "C" void kernel_launch(void* const* d_in, const int* in_sizes, int n_in,
                              void* d_out, int out_size, void* d_ws,
                              size_t ws_size, hipStream_t stream) {
  (void)in_sizes; (void)n_in; (void)out_size; (void)ws_size;
  const int* eidx = (const int*)d_in[0];
  const float* wgt = (const float*)d_in[1];
  const float* rps = (const float*)d_in[2];
  const float* rrt = (const float*)d_in[3];
  const float* dlt = (const float*)d_in[4];
  /* d_in[5] mask: unused by reference */
  const float* exe = (const float*)d_in[6];
  const float* mue = (const float*)d_in[7];
  const float* ltau = (const float*)d_in[8];
  const float* inv = (const float*)d_in[9];
  const float* fW1 = (const float*)d_in[10];
  const float* fb1 = (const float*)d_in[11];
  const float* fW2 = (const float*)d_in[12];
  const float* fb2 = (const float*)d_in[13];
  const float* fW3 = (const float*)d_in[14];
  const float* fb3 = (const float*)d_in[15];
  const float* fW4 = (const float*)d_in[16];
  const float* fb4 = (const float*)d_in[17];
  const float* gW1 = (const float*)d_in[18];
  const float* gb1 = (const float*)d_in[19];
  const float* gW2 = (const float*)d_in[20];
  const float* gb2 = (const float*)d_in[21];
  const float* gW3 = (const float*)d_in[22];
  const float* gb3 = (const float*)d_in[23];
  const float* gW4 = (const float*)d_in[24];
  const float* gb4 = (const float*)d_in[25];

  char* ws = (char*)d_ws;
  float* mpcPost = (float*)(ws + WS_MPC);
  float* fE = (float*)(ws + WS_FE);
  float* fMuB = (float*)(ws + WS_FMU);
  float* gE = (float*)(ws + WS_GE);
  _Float16* w2frag = (_Float16*)(ws + WS_W2F);
  _Float16* w3frag = (_Float16*)(ws + WS_W3F);
  _Float16* gw2frag = (_Float16*)(ws + WS_GW2F);
  _Float16* gw3frag = (_Float16*)(ws + WS_GW3F);

  float* outRir = (float*)d_out;              // (B,T)
  float* outMpc = outRir + BB * TT;           // (B,M)

  precompute_tables<<<NEX + MM + NEX, 256, 0, stream>>>(exe, mue, fW1, fb1, gW1,
                                                        gb1, fE, fMuB, gE);
  swizzle_bfrag<<<8 * 16, 256, 0, stream>>>(fW2, w2frag, 16, 256);
  swizzle_bfrag<<<8 * 8, 256, 0, stream>>>(fW3, w3frag, 8, 128);
  swizzle_bfrag<<<8 * 16, 256, 0, stream>>>(gW2, gw2frag, 16, 256);
  swizzle_bfrag<<<8 * 8, 256, 0, stream>>>(gW3, gw3frag, 8, 128);
  scan_kernel<<<BB / 4, 256, SMEM_B, stream>>>(
      eidx, wgt, rps, rrt, dlt, ltau, inv, fb2, fb3, fW4, fb4, fW1, fE, fMuB,
      w2frag, w3frag, mpcPost, outMpc);
  rir_kernel<<<(BB * TT) / 64, 256, SMEM_C, stream>>>(
      eidx, wgt, rps, gW1, gb2, gb3, gW4, gb4, gE, gw2frag, gw3frag, mpcPost,
      outRir);
}
